// GlobalLocalLoss_23407571763381
// MI455X (gfx1250) — matrix operand order repaired and checked by reading, not compiled, not run on hardware
//
#include <hip/hip_runtime.h>

// ---------------------------------------------------------------------------
// MI455X (gfx1250, wave32) implementation of the global+local RWA loss.
// All heavy matmuls go through v_wmma_f32_16x16x32_bf16 (f32 accumulate).
// ---------------------------------------------------------------------------

typedef __attribute__((ext_vector_type(16))) __bf16 bf16x16;
typedef __attribute__((ext_vector_type(8)))  __bf16 bf16x8;
typedef __attribute__((ext_vector_type(8)))  float  fx8;

__device__ __forceinline__ fx8 wmma_bf16(bf16x16 a, bf16x16 b, fx8 c) {
    // 8 args: (neg_a, A, neg_b, B, c_mod, C, reuse_a, reuse_b)
    return __builtin_amdgcn_wmma_f32_16x16x32_bf16(
        false, a, false, b, (short)0, c, false, false);
}

__device__ __forceinline__ bf16x8 ld8(const __bf16* p) { return *(const bf16x8*)p; }

__device__ __forceinline__ bf16x16 cat8(bf16x8 lo, bf16x8 hi) {
    bf16x16 r;
#pragma unroll
    for (int i = 0; i < 8; ++i) { r[i] = lo[i]; r[i + 8] = hi[i]; }
    return r;
}

__device__ __forceinline__ fx8 zero8() {
    fx8 z;
#pragma unroll
    for (int i = 0; i < 8; ++i) z[i] = 0.0f;
    return z;
}

// --------------------------- block reductions ------------------------------
__device__ __forceinline__ float blk_sum(float* red, int tid, float v) {
    red[tid] = v; __syncthreads();
#pragma unroll
    for (int s = 64; s > 0; s >>= 1) {
        if (tid < s) red[tid] += red[tid + s];
        __syncthreads();
    }
    float r = red[0]; __syncthreads();
    return r;
}
__device__ __forceinline__ float blk_max(float* red, int tid, float v) {
    red[tid] = v; __syncthreads();
#pragma unroll
    for (int s = 64; s > 0; s >>= 1) {
        if (tid < s) red[tid] = fmaxf(red[tid], red[tid + s]);
        __syncthreads();
    }
    float r = red[0]; __syncthreads();
    return r;
}

// ---------------------------------------------------------------------------
// Kernel 0: per-row l2 normalization of im (B*64,512) and cap (B*32,512).
// Emits bf16 normalized rows + exact f32 row norms.
// ---------------------------------------------------------------------------
__global__ __launch_bounds__(128)
void rownorm_kernel(const float* __restrict__ im, const float* __restrict__ cap,
                    __bf16* __restrict__ imn, __bf16* __restrict__ capn,
                    float* __restrict__ imrn, float* __restrict__ caprn,
                    int nImRows) {
    __shared__ float red[128];
    const int b = blockIdx.x, tid = threadIdx.x;
    const float* src; __bf16* dst; float* rn;
    if (b < nImRows) { src = im  + (size_t)b * 512; dst = imn  + (size_t)b * 512; rn = imrn  + b; }
    else { int i = b - nImRows; src = cap + (size_t)i * 512; dst = capn + (size_t)i * 512; rn = caprn + i; }
    float4 v = ((const float4*)src)[tid];
    float s = v.x * v.x + v.y * v.y + v.z * v.z + v.w * v.w;
    float tot = blk_sum(red, tid, s);
    float nrm = sqrtf(tot);
    float inv = 1.0f / (nrm + 1e-8f);
    dst[tid * 4 + 0] = (__bf16)(v.x * inv);
    dst[tid * 4 + 1] = (__bf16)(v.y * inv);
    dst[tid * 4 + 2] = (__bf16)(v.z * inv);
    dst[tid * 4 + 3] = (__bf16)(v.w * inv);
    if (tid == 0) *rn = nrm;
}

// ---------------------------------------------------------------------------
// Kernel 1/2: one (ctx batch c, qry batch q) pair per block, 4 waves.
//   S[ls][lq]   = ctx_norm[c] . qry_norm[q]^T           (WMMA, bf16)
//   leaky -> l2norm over lq -> +masks -> softmax over ls -> P[lq][ls] (bf16)
//   W[lq][f]    = P . ctx_norm[c]                        (WMMA, bf16)
//   cos(qry raw row, W row) fused into GEMM2 epilogue; mean over lq.
// ---------------------------------------------------------------------------
template<int LS, int LQ>
__global__ __launch_bounds__(128)
void attn_pair(const __bf16* __restrict__ ctxn, const __bf16* __restrict__ qryn,
               const float*  __restrict__ qrn,  const float* __restrict__ qmask,
               const float*  __restrict__ cmask, float* __restrict__ simOut, int B) {
    constexpr int NT   = LQ / 16;       // GEMM1 N tiles
    constexpr int MT2  = LQ / 16;       // GEMM2 M tiles
    constexpr int KT2  = LS / 32;       // GEMM2 K steps
    constexpr int QS   = 520;           // qry LDS stride (bf16), bank-friendly
    constexpr int CTS  = LS + 8;        // ctxT stride (bf16)
    constexpr int SST  = LQ + 4;        // S stride (f32)
    constexpr int PSTB = LS + 16;       // P stride (bf16), 16B-aligned rows

    __shared__ __align__(16) __bf16 qryL[LQ * QS];     // qry_norm rows (GEMM1 B, cosine)
    __shared__ __align__(16) __bf16 ctxT[512 * CTS];   // ctx_norm transposed (GEMM2 B)
    __shared__ __align__(16) float  Sb[LS * SST];      // raw/normalized scores
    __shared__ __align__(16) __bf16 Pb[LQ * PSTB];     // attention probs, WMMA-A layout
    __shared__ float mqA[LQ], mcA[LS];
    __shared__ float s12W[4][LQ], s22W[4][LQ];         // per-wave cosine partials
    __shared__ float red[128];

    const int tid  = threadIdx.x;
    const int wave = tid >> 5, lane = tid & 31;
    const int half = lane >> 4, l16 = lane & 15;
    const int q = blockIdx.x, c = blockIdx.y;

    const __bf16* cg = ctxn + (size_t)c * LS * 512;
    const __bf16* qg = qryn + (size_t)q * LQ * 512;

    // ---- stage qry rows (row-major) and ctx (transposed) into LDS ----
    for (int ch = tid; ch < LQ * 64; ch += 128) {
        int r = ch >> 6, o = (ch & 63) << 3;
        *(bf16x8*)&qryL[r * QS + o] = ld8(qg + r * 512 + o);
    }
    for (int ch = tid; ch < LS * 64; ch += 128) {
        int r = ch >> 6, o = (ch & 63) << 3;
        bf16x8 v = ld8(cg + r * 512 + o);
#pragma unroll
        for (int i = 0; i < 8; ++i) ctxT[(o + i) * CTS + r] = v[i];
    }
    for (int t = tid; t < LQ; t += 128) mqA[t] = qmask[(size_t)q * LQ + t];
    for (int t = tid; t < LS; t += 128) mcA[t] = cmask[(size_t)c * LS + t];
    __syncthreads();

    // ---- GEMM1: S = ctx . qry^T, tiles {2w, 2w+1} per wave ----
    {
        const int t0  = 2 * wave;
        const int mt  = t0 / NT;
        const int nt0 = t0 % NT;                 // nt1 = nt0+1 (NT is even)
        fx8 acc0 = zero8(), acc1 = zero8();
        const __bf16* arow = cg + (size_t)(mt * 16 + l16) * 512;
        __builtin_prefetch(arow, 0, 1);          // global_prefetch_b8
#pragma unroll
        for (int kk = 0; kk < 16; ++kk) {        // K = 512 in steps of 32
            const int K = kk * 32;
            bf16x16 a = cat8(ld8(arow + K + half * 8),
                             ld8(arow + K + 16 + half * 8));
            const __bf16* b0p = &qryL[(nt0 * 16 + l16) * QS + K + half * 8];
            const __bf16* b1p = &qryL[((nt0 + 1) * 16 + l16) * QS + K + half * 8];
            bf16x16 b0 = cat8(ld8(b0p), ld8(b0p + 16));
            bf16x16 b1 = cat8(ld8(b1p), ld8(b1p + 16));
            acc0 = wmma_bf16(a, b0, acc0);
            acc1 = wmma_bf16(a, b1, acc1);
        }
        // LeakyReLU(0.1) while writing C tiles (VGPR r: M=r / r+8, N=l16)
#pragma unroll
        for (int r = 0; r < 8; ++r) {
            const int m = mt * 16 + (half ? r + 8 : r);
            float v0 = acc0[r]; v0 = v0 > 0.0f ? v0 : 0.1f * v0;
            Sb[m * SST + nt0 * 16 + l16] = v0;
            float v1 = acc1[r]; v1 = v1 > 0.0f ? v1 : 0.1f * v1;
            Sb[m * SST + (nt0 + 1) * 16 + l16] = v1;
        }
    }
    __syncthreads();

    // ---- l2norm over lq within each ls row ----
    for (int r = tid; r < LS; r += 128) {
        float s = 0.0f;
        for (int n = 0; n < LQ; ++n) { float v = Sb[r * SST + n]; s += v * v; }
        float inv = 1.0f / (sqrtf(s) + 1e-8f);
        for (int n = 0; n < LQ; ++n) Sb[r * SST + n] *= inv;
    }
    __syncthreads();

    // ---- masks + lambda softmax over ls, write P[lq][ls] as bf16 ----
    for (int n = tid; n < LQ; n += 128) {
        const float mq = mqA[n];
        float mx = -3.4e38f;
        for (int r = 0; r < LS; ++r)
            mx = fmaxf(mx, 20.0f * (Sb[r * SST + n] + mq + mcA[r]));
        float Z = 0.0f;
        for (int r = 0; r < LS; ++r) {
            float e = expf(20.0f * (Sb[r * SST + n] + mq + mcA[r]) - mx);
            Pb[n * PSTB + r] = (__bf16)e;        // unnormalized, rescaled below
            Z += e;
        }
        float iz = 1.0f / Z;
        for (int r = 0; r < LS; ++r)
            Pb[n * PSTB + r] = (__bf16)((float)Pb[n * PSTB + r] * iz);
    }
    __syncthreads();

    // ---- GEMM2: W = P . ctx, cosine partials fused into epilogue ----
    for (int mt = 0; mt < MT2; ++mt) {
        float s12r[8], s22r[8];
#pragma unroll
        for (int r = 0; r < 8; ++r) { s12r[r] = 0.0f; s22r[r] = 0.0f; }
        const __bf16* prow = &Pb[(mt * 16 + l16) * PSTB];
        for (int nt = wave; nt < 32; nt += 4) {  // 512 features / 16 per tile
            fx8 acc = zero8();
#pragma unroll
            for (int kk = 0; kk < KT2; ++kk) {
                const int K = kk * 32;
                bf16x16 a = cat8(ld8(prow + K + half * 8),
                                 ld8(prow + K + 16 + half * 8));
                const __bf16* bp = &ctxT[(nt * 16 + l16) * CTS + K + half * 8];
                bf16x16 b = cat8(ld8(bp), ld8(bp + 16));
                acc = wmma_bf16(a, b, acc);
            }
            const int f = nt * 16 + l16;
#pragma unroll
            for (int r = 0; r < 8; ++r) {
                const int m = mt * 16 + (half ? r + 8 : r);
                float v  = acc[r];
                float qv = (float)qryL[m * QS + f];
                s12r[r] += qv * v;       // qry_norm . W
                s22r[r] += v * v;        // ||W||^2
            }
        }
        // reduce over the 16 lanes that share each row m
#pragma unroll
        for (int r = 0; r < 8; ++r) {
            float a12 = s12r[r], a22 = s22r[r];
#pragma unroll
            for (int msk = 1; msk < 16; msk <<= 1) {
                a12 += __shfl_xor(a12, msk, 32);
                a22 += __shfl_xor(a22, msk, 32);
            }
            if (l16 == 0) {
                const int m = mt * 16 + half * 8 + r;
                s12W[wave][m] = a12;
                s22W[wave][m] = a22;
            }
        }
    }
    __syncthreads();

    // ---- cosine + mean over lq ----
    float mycos = 0.0f;
    if (tid < LQ) {
        float d12 = s12W[0][tid] + s12W[1][tid] + s12W[2][tid] + s12W[3][tid];
        float d22 = s22W[0][tid] + s22W[1][tid] + s22W[2][tid] + s22W[3][tid];
        float w1  = qrn[(size_t)q * LQ + tid];           // exact ||qry row||
        float w12 = (w1 + 1e-8f) * d12;                  // raw = norm*(w1+eps)
        float w2  = sqrtf(d22);
        mycos = w12 / fmaxf(w1 * w2, 1e-8f);
    }
    float csum = blk_sum(red, tid, mycos);
    if (tid == 0) simOut[(size_t)c * B + q] = csum / (float)LQ;
}

// ---------------------------------------------------------------------------
// Kernel 3: per-row global-loss terms and local KL-row losses. B == 128.
// ---------------------------------------------------------------------------
__global__ __launch_bounds__(128)
void score_loss_kernel(const float* __restrict__ gs, const float* __restrict__ i2t,
                       const float* __restrict__ t2i, float* __restrict__ rowterm,
                       float* __restrict__ localrow, int B) {
    __shared__ float red[128];
    const int i = blockIdx.x, j = threadIdx.x;
    const float invT = 20.0f;   // 1/0.05

    // log-sum-exp of row i of gs/T
    float xr = gs[(size_t)i * B + j] * invT;
    float mr = blk_max(red, j, xr);
    float Zr = blk_sum(red, j, expf(xr - mr));
    float lse_r = mr + logf(Zr);

    // log-sum-exp of column i of gs/T
    float xc = gs[(size_t)j * B + i] * invT;
    float mc = blk_max(red, j, xc);
    float Zc = blk_sum(red, j, expf(xc - mc));
    float lse_c = mc + logf(Zc);

    // local loss row i: scores[i][j] = i2t[i][j] + t2i[j][i]
    float sc = i2t[(size_t)i * B + j] + t2i[(size_t)j * B + i];
    float l  = sc * 20.0f;
    float ml = blk_max(red, j, l);
    float el = expf(l - ml);
    float Zl = blk_sum(red, j, el);
    float logsm = l - ml - logf(Zl);
    float p     = el / Zl;
    float lab   = logf(((j == i) ? 1.0f : 0.0f) + 1e-6f);
    float lsum  = blk_sum(red, j, p * (logsm - lab));

    if (j == 0) {
        rowterm[i]  = 2.0f * (gs[(size_t)i * B + i] * invT) - lse_r - lse_c;
        localrow[i] = lsum;
    }
}

// ---------------------------------------------------------------------------
// Kernel 4: finalize -> (loss, global_loss, local_loss)
// ---------------------------------------------------------------------------
__global__ __launch_bounds__(128)
void finalize_kernel(const float* __restrict__ rowterm, const float* __restrict__ localrow,
                     float* __restrict__ out, int B) {
    __shared__ float red[128];
    const int tid = threadIdx.x;
    float a = (tid < B) ? rowterm[tid]  : 0.0f;
    float b = (tid < B) ? localrow[tid] : 0.0f;
    float s1 = blk_sum(red, tid, a);
    float s2 = blk_sum(red, tid, b);
    if (tid == 0) {
        float gl = -s1 / (float)B;
        float lo =  s2 / (float)B;
        out[0] = gl + lo;
        out[1] = gl;
        out[2] = lo;
    }
}

// ---------------------------------------------------------------------------
extern "C" void kernel_launch(void* const* d_in, const int* in_sizes, int n_in,
                              void* d_out, int out_size, void* d_ws, size_t ws_size,
                              hipStream_t stream) {
    const float* gs   = (const float*)d_in[0];   // (B,B)
    const float* im   = (const float*)d_in[1];   // (B,64,512)
    const float* cap  = (const float*)d_in[2];   // (B,32,512)
    const float* im_m = (const float*)d_in[3];   // (B,64)
    /* d_in[4] local_s_l unused in the fast path */
    const float* s_m  = (const float*)d_in[5];   // (B,32)

    const int B  = in_sizes[4];                  // 128
    const int Lr = 64, Lw = 32;

    char* w = (char*)d_ws;
    __bf16* imn   = (__bf16*)w; w += (size_t)B * Lr * 512 * 2;
    __bf16* capn  = (__bf16*)w; w += (size_t)B * Lw * 512 * 2;
    float*  imrn  = (float*)w;  w += (size_t)B * Lr * 4;
    float*  caprn = (float*)w;  w += (size_t)B * Lw * 4;
    float*  i2t   = (float*)w;  w += (size_t)B * B * 4;
    float*  t2i   = (float*)w;  w += (size_t)B * B * 4;
    float*  rowt  = (float*)w;  w += (size_t)B * 4;
    float*  locr  = (float*)w;  w += (size_t)B * 4;

    rownorm_kernel<<<B * (Lr + Lw), 128, 0, stream>>>(im, cap, imn, capn, imrn, caprn, B * Lr);

    // i2t: ctx = images (Ls=64), qry = captions (Lq=32)
    attn_pair<64, 32><<<dim3(B, B), 128, 0, stream>>>(imn, capn, caprn, s_m, im_m, i2t, B);
    // t2i: ctx = captions (Ls=32), qry = images (Lq=64)
    attn_pair<32, 64><<<dim3(B, B), 128, 0, stream>>>(capn, imn, imrn, im_m, s_m, t2i, B);

    score_loss_kernel<<<B, 128, 0, stream>>>(gs, i2t, t2i, rowt, locr, B);
    finalize_kernel<<<1, 128, 0, stream>>>(rowt, locr, (float*)d_out, B);
}
